// SuperNet_8967891714119
// MI455X (gfx1250) — matrix operand order repaired and checked by reading, not compiled, not run on hardware
//
#include <hip/hip_runtime.h>
#include <hip/hip_bf16.h>

#define NFEAT  256
#define HIDDEN 64
#define NCLASS 40

typedef __bf16 bf16_t;
typedef __bf16 bf16x8  __attribute__((ext_vector_type(8)));
typedef __bf16 bf16x16 __attribute__((ext_vector_type(16)));
typedef float  f32x8   __attribute__((ext_vector_type(8)));

// ---------------------------------------------------------------------------
// WMMA fragment loaders (CDNA5 16x16x32 bf16 layouts, ISA 7.12.2)
// A (16x32, bf16): lane l -> M = l&15 ; klane = (l>=16)*8 ;
//   fragment elems 0..7  = A[m][k0+klane+0..7]   (contiguous)
//   fragment elems 8..15 = A[m][k0+klane+16..23] (contiguous)
// ---------------------------------------------------------------------------
__device__ __forceinline__ bf16x16 load_a_f32(const float* __restrict__ A,
                                              int lda, int row0, int k0) {
  const int l = threadIdx.x & 31;
  const float* base = A + (size_t)(row0 + (l & 15)) * lda + k0 + ((l >> 4) << 3);
  const float4 v0 = *(const float4*)(base);       // k + 0..3
  const float4 v1 = *(const float4*)(base + 4);   // k + 4..7
  const float4 v2 = *(const float4*)(base + 16);  // k + 16..19
  const float4 v3 = *(const float4*)(base + 20);  // k + 20..23
  bf16x16 a;
  a[0] = (bf16_t)v0.x;  a[1] = (bf16_t)v0.y;  a[2]  = (bf16_t)v0.z;  a[3]  = (bf16_t)v0.w;
  a[4] = (bf16_t)v1.x;  a[5] = (bf16_t)v1.y;  a[6]  = (bf16_t)v1.z;  a[7]  = (bf16_t)v1.w;
  a[8] = (bf16_t)v2.x;  a[9] = (bf16_t)v2.y;  a[10] = (bf16_t)v2.z;  a[11] = (bf16_t)v2.w;
  a[12] = (bf16_t)v3.x; a[13] = (bf16_t)v3.y; a[14] = (bf16_t)v3.z;  a[15] = (bf16_t)v3.w;
  return a;
}

__device__ __forceinline__ bf16x16 load_a_bf16(const bf16_t* __restrict__ A,
                                               int lda, int row0, int k0) {
  const int l = threadIdx.x & 31;
  const bf16_t* base = A + (size_t)(row0 + (l & 15)) * lda + k0 + ((l >> 4) << 3);
  const bf16x8 lo = *(const bf16x8*)(base);       // k + 0..7   (16B load)
  const bf16x8 hi = *(const bf16x8*)(base + 16);  // k + 16..23 (16B load)
  return __builtin_shufflevector(lo, hi, 0, 1, 2, 3, 4, 5, 6, 7, 8, 9, 10, 11,
                                 12, 13, 14, 15);
}

// B fragments are pre-swizzled in LDS: fragment (kb, nt) for lane l lives at
// 32B-aligned offset ((kb*NT + nt)*32 + l)*16 elements -> one 32B vector load.
__device__ __forceinline__ bf16x16 load_b_swz(const bf16_t* __restrict__ Wl,
                                              int fragIdx) {
  const int l = threadIdx.x & 31;
  return *(const bf16x16*)(Wl + ((size_t)fragIdx * 32 + l) * 16);
}

// ---------------------------------------------------------------------------
// GEMM: Out[nrows x NOUT] = act(A[nrows x K] @ W[K x NOUT] (+ bias))
// One wave computes a 16 x NC output tile; weights staged swizzled in LDS.
// B layout (32x16 bf16): lane l -> N = l&15 ; lanes 0-15 hold K 0..7,16..23,
// lanes 16-31 hold K 8..15,24..31 ; elem e -> K = klane + (e&7) + (e>>3)*16.
// ---------------------------------------------------------------------------
template <int K, int NC, int NOUT, bool AF32, bool ACT, bool OUTBF16>
__global__ void __launch_bounds__(256)
gemm_wmma_kernel(const void* __restrict__ Asrc, const float* __restrict__ Wg,
                 const float* __restrict__ bias, void* __restrict__ Out,
                 int nrows) {
  constexpr int NT = NC / 16;
  constexpr int KB = K / 32;
  __shared__ bf16_t Wl[K * NC];  // fragment-major swizzled
  for (int idx = threadIdx.x; idx < K * NC; idx += blockDim.x) {
    const int frag = idx >> 9;         // / 512 elems per fragment
    const int lane = (idx >> 4) & 31;
    const int e = idx & 15;
    const int kb = frag / NT, nt = frag % NT;
    const int k = kb * 32 + ((lane >> 4) << 3) + (e & 7) + ((e >> 3) << 4);
    const int n = nt * 16 + (lane & 15);
    const float v = (n < NOUT) ? Wg[k * NOUT + n] : 0.0f;  // zero-pad cols
    Wl[idx] = (bf16_t)v;
  }
  __syncthreads();

  const int wavesPerBlk = blockDim.x >> 5;
  const int ntiles = nrows >> 4;  // nrows % 16 == 0 (N = 100000)

  for (int tile = blockIdx.x * wavesPerBlk + (threadIdx.x >> 5); tile < ntiles;
       tile += gridDim.x * wavesPerBlk) {
    const int row0 = tile << 4;
    f32x8 acc[NT] = {};
#pragma unroll
    for (int kb = 0; kb < KB; ++kb) {
      bf16x16 a;
      if (AF32) a = load_a_f32((const float*)Asrc, K, row0, kb * 32);
      else      a = load_a_bf16((const bf16_t*)Asrc, K, row0, kb * 32);
#pragma unroll
      for (int t = 0; t < NT; ++t) {
        bf16x16 b = load_b_swz(Wl, kb * NT + t);
        acc[t] = __builtin_amdgcn_wmma_f32_16x16x32_bf16(
            false, a, false, b, (short)0, acc[t], false, false);
      }
    }
    // Epilogue: C/D layout -> lane l, VGPR r holds (M = r + (l>=16)*8, N = l&15)
    const int l = threadIdx.x & 31;
    const int mbase = row0 + ((l >> 4) << 3);
#pragma unroll
    for (int t = 0; t < NT; ++t) {
      const int n = t * 16 + (l & 15);
      if (n < NOUT) {
        const float bv = ACT ? bias[n] : 0.0f;
#pragma unroll
        for (int r = 0; r < 8; ++r) {
          float v = acc[t][r];
          if (ACT) { v += bv; v = v > 0.0f ? v : 0.0f; }
          const size_t off = (size_t)(mbase + r) * NOUT + n;
          if (OUTBF16) ((bf16_t*)Out)[off] = (bf16_t)v;
          else         ((float*)Out)[off]  = v;
        }
      }
    }
  }
}

// ---------------------------------------------------------------------------
// Graph kernels
// ---------------------------------------------------------------------------
__global__ void init_kernel(float* __restrict__ deg, float* __restrict__ lsum,
                            int n_nodes) {
  const int t = blockIdx.x * blockDim.x + threadIdx.x;
  if (t < n_nodes * HIDDEN) lsum[t] = 0.0f;
  if (t < n_nodes) deg[t] = 1.0f;  // self-loop contribution
}

__global__ void degree_kernel(const int* __restrict__ dst,
                              float* __restrict__ deg, int nedges) {
  const int e = blockIdx.x * blockDim.x + threadIdx.x;
  if (e < nedges) unsafeAtomicAdd(&deg[dst[e]], 1.0f);
}

__global__ void rsqrt_kernel(const float* __restrict__ deg,
                             float* __restrict__ dinv, int n_nodes) {
  const int i = blockIdx.x * blockDim.x + threadIdx.x;
  if (i < n_nodes) dinv[i] = rsqrtf(deg[i]);  // deg >= 1 always
}

// One thread per (edge, 16-float chunk): gather hw[src]*norm, scatter-add to
// agg[dst]. hw/agg tables are 25.6 MB each -> L2-resident (192 MB L2), so the
// random phase runs at L2 bandwidth with f32 atomics resolved in L2.
__global__ void scatter_kernel(const int* __restrict__ src,
                               const int* __restrict__ dst,
                               const float* __restrict__ dinv,
                               const float* __restrict__ hw,
                               float* __restrict__ agg, int nedges) {
  const int idx = blockIdx.x * blockDim.x + threadIdx.x;
  const int e = idx >> 2;
  if (e >= nedges) return;
  const int c = (idx & 3) << 4;
  const int s = src[e], d = dst[e];
  const float nm = dinv[s] * dinv[d];
  const float4* hp = (const float4*)(hw + (size_t)s * HIDDEN + c);
  float* ap = agg + (size_t)d * HIDDEN + c;
#pragma unroll
  for (int q = 0; q < 4; ++q) {
    const float4 v = hp[q];
    unsafeAtomicAdd(ap + 4 * q + 0, v.x * nm);
    unsafeAtomicAdd(ap + 4 * q + 1, v.y * nm);
    unsafeAtomicAdd(ap + 4 * q + 2, v.z * nm);
    unsafeAtomicAdd(ap + 4 * q + 3, v.w * nm);
  }
}

// l = agg + self_loop + bias ; lsum += l ; next-layer activations re-quantized
// to bf16 (last layer emits z/3 so the head GEMM consumes it directly).
__global__ void finish_kernel(const float* __restrict__ agg,
                              const float* __restrict__ hw,
                              const float* __restrict__ dinv,
                              const float* __restrict__ bias,
                              float* __restrict__ lsum,
                              bf16_t* __restrict__ hout, int n_nodes, int last) {
  const int t = blockIdx.x * blockDim.x + threadIdx.x;
  if (t >= n_nodes * HIDDEN) return;
  const int i = t >> 6, f = t & 63;
  const float di = dinv[i];
  const float v = agg[t] + hw[t] * di * di + bias[f];
  const float s = lsum[t] + v;
  lsum[t] = s;
  hout[t] = (bf16_t)(last ? s * (1.0f / 3.0f) : v);
}

// ---------------------------------------------------------------------------
extern "C" void kernel_launch(void* const* d_in, const int* in_sizes, int n_in,
                              void* d_out, int out_size, void* d_ws,
                              size_t ws_size, hipStream_t stream) {
  const float* x  = (const float*)d_in[0];
  const int*   ei = (const int*)d_in[1];
  const float* Wx = (const float*)d_in[2];
  const float* bx = (const float*)d_in[3];
  const float* Wgcn[3] = {(const float*)d_in[4], (const float*)d_in[6],
                          (const float*)d_in[8]};
  const float* bgcn[3] = {(const float*)d_in[5], (const float*)d_in[7],
                          (const float*)d_in[9]};
  const float* Wz = (const float*)d_in[10];
  const float* bz = (const float*)d_in[11];
  float* out = (float*)d_out;
  (void)n_in; (void)out_size; (void)ws_size;

  const int N = in_sizes[0] / NFEAT;
  const int E = in_sizes[1] / 2;
  const int* src = ei;
  const int* dst = ei + E;

  // Workspace carve (≈ 90.4 MB total)
  char* ws = (char*)d_ws;
  size_t off = 0;
  auto carve = [&](size_t bytes) -> void* {
    void* p = (void*)(ws + off);
    off += (bytes + 255) & ~(size_t)255;
    return p;
  };
  float*  deg  = (float*)carve((size_t)N * sizeof(float));
  float*  dinv = (float*)carve((size_t)N * sizeof(float));
  bf16_t* hbf  = (bf16_t*)carve((size_t)N * HIDDEN * sizeof(bf16_t));
  float*  hw   = (float*)carve((size_t)N * HIDDEN * sizeof(float));
  float*  agg  = (float*)carve((size_t)N * HIDDEN * sizeof(float));
  float*  lsum = (float*)carve((size_t)N * HIDDEN * sizeof(float));

  const int tot = N * HIDDEN;
  init_kernel<<<(tot + 255) / 256, 256, 0, stream>>>(deg, lsum, N);
  degree_kernel<<<(E + 255) / 256, 256, 0, stream>>>(dst, deg, E);
  rsqrt_kernel<<<(N + 255) / 256, 256, 0, stream>>>(deg, dinv, N);

  const int ntiles = N / 16;                 // N = 100000 -> 6250 tiles
  const int gemmBlocks = (ntiles + 7) / 8;   // 8 waves per 256-thread block

  // h0 = relu(x @ Wx + bx), stored bf16 for next GEMM
  gemm_wmma_kernel<NFEAT, HIDDEN, HIDDEN, true, true, true>
      <<<gemmBlocks, 256, 0, stream>>>(x, Wx, bx, hbf, N);

  for (int layer = 0; layer < 3; ++layer) {
    hipMemsetAsync(agg, 0, (size_t)N * HIDDEN * sizeof(float), stream);
    gemm_wmma_kernel<HIDDEN, HIDDEN, HIDDEN, false, false, false>
        <<<gemmBlocks, 256, 0, stream>>>(hbf, Wgcn[layer], nullptr, hw, N);
    scatter_kernel<<<(E * 4 + 255) / 256, 256, 0, stream>>>(src, dst, dinv, hw,
                                                            agg, E);
    finish_kernel<<<(tot + 255) / 256, 256, 0, stream>>>(
        agg, hw, dinv, bgcn[layer], lsum, hbf, N, layer == 2);
  }

  // out = relu((z/3) @ Wz + bz), 40 cols padded to 48 (3 WMMA n-tiles)
  gemm_wmma_kernel<HIDDEN, 48, NCLASS, false, true, false>
      <<<gemmBlocks, 256, 0, stream>>>(hbf, Wz, bz, out, N);
}
